// AttentionPointerCoverageDecoder_35373350650615
// MI455X (gfx1250) — compile-verified, matrix operand order checked by reading, not run on hardware
//
#include <hip/hip_runtime.h>
#include <hip/hip_bf16.h>

// Problem constants
#define NB   128      // B
#define NT   400      // T
#define NE   128      // E
#define NV   50000    // V
#define NH2  256      // H2
#define NH4  512      // H4

typedef __attribute__((ext_vector_type(16))) __bf16 v16bf;
typedef __attribute__((ext_vector_type(8)))  float  v8f;
typedef __attribute__((ext_vector_type(4)))  unsigned int v4u;
typedef __attribute__((ext_vector_type(8)))  int    v8i;
typedef __attribute__((ext_vector_type(4)))  int    v4i;

// ---------------- workspace layout (float offsets into d_ws) ----------------
#define WS_GATES   0                       // 128*1024
#define WS_H       131072                  // 128*256
#define WS_CTX     163840                  // 128*256
#define WS_SCORES  196608                  // 128*400
#define WS_SH      247808                  // 128
#define WS_CONST   247936                  // 2: s_cov, s_b
#define WS_PGEN    247940                  // 128
#define WS_RMAX    248068                  // 128
#define WS_RSUM    248196                  // 128
#define WS_FCB     248324                  // bf16 128*512 (occupies 32768 floats)

// ---------------- output layout (float offsets into d_out) -----------------
#define O_OUT   0                          // (B,V)
#define O_H     6400000                    // (1,B,H2)
#define O_C     6432768                    // (1,B,H2)
#define O_ATT   6465536                    // (B,1,T)
#define O_COPY  6516736                    // (B,V)
#define O_NCOV  12916736                   // (B,1,T)
#define O_CL    12967936                   // scalar

__device__ __forceinline__ float waveReduceSum(float v) {
  #pragma unroll
  for (int m = 16; m >= 1; m >>= 1) v += __shfl_xor(v, m, 32);
  return v;
}

__device__ __forceinline__ void atomicMaxFloat(float* addr, float v) {
  unsigned int* ai = (unsigned int*)addr;
  unsigned int old = *ai;
  while (true) {
    float f = __uint_as_float(old);
    if (f >= v) break;
    unsigned int assumed = old;
    old = atomicCAS(ai, assumed, __float_as_uint(v));
    if (old == assumed) break;
  }
}

// ---------------------------------------------------------------------------
__global__ void k_init(float* ws, float* out) {
  int t = threadIdx.x;
  if (t < NB) { ws[WS_RMAX + t] = -3.0e38f; ws[WS_RSUM + t] = 0.f; }
  if (t == 0) out[O_CL] = 0.f;
}

// LSTM gates: gates[b][j] = bias + emb[b]·w_ih[j] + h0[b]·w_hh[j]
// grid(16) block(256); 8 batch rows per block, fp32 FMA (tiny GEMM, exact).
__global__ void k_gates(const int* __restrict__ ids,
                        const float* __restrict__ emb_table,
                        const float* __restrict__ h0,
                        const float* __restrict__ w_ih,
                        const float* __restrict__ w_hh,
                        const float* __restrict__ b_ih,
                        const float* __restrict__ b_hh,
                        float* __restrict__ ws) {
  __shared__ float xs[8][384];           // emb(128) | h0(256) per row
  const int bb = blockIdx.x * 8;
  const int tid = threadIdx.x;
  for (int i = tid; i < 8 * 384; i += 256) {
    int lb = i / 384, e = i % 384, b = bb + lb;
    xs[lb][e] = (e < 128) ? emb_table[(size_t)ids[b] * NE + e]
                          : h0[b * NH2 + (e - 128)];
  }
  __syncthreads();
  for (int q = 0; q < 4; ++q) {
    int j = tid + 256 * q;               // 0..1023
    float acc[8];
    float bias = b_ih[j] + b_hh[j];
    #pragma unroll
    for (int lb = 0; lb < 8; ++lb) acc[lb] = bias;
    const float* wi = w_ih + (size_t)j * NE;
    for (int k = 0; k < NE; ++k) {
      float w = wi[k];
      #pragma unroll
      for (int lb = 0; lb < 8; ++lb) acc[lb] += w * xs[lb][k];
    }
    const float* wh = w_hh + (size_t)j * NH2;
    for (int k = 0; k < NH2; ++k) {
      float w = wh[k];
      #pragma unroll
      for (int lb = 0; lb < 8; ++lb) acc[lb] += w * xs[lb][128 + k];
    }
    #pragma unroll
    for (int lb = 0; lb < 8; ++lb) ws[WS_GATES + (bb + lb) * 1024 + j] = acc[lb];
  }
}

// LSTM pointwise. grid(128) block(256)
__global__ void k_lstm(const float* __restrict__ c0p, float* ws, float* out) {
  int b = blockIdx.x, j = threadIdx.x;
  const float* g = ws + WS_GATES + b * 1024;
  float i = g[j], f = g[256 + j], gg = g[512 + j], o = g[768 + j];
  float c0 = c0p[b * NH2 + j];
  float si = 1.f / (1.f + expf(-i));
  float sf = 1.f / (1.f + expf(-f));
  float so = 1.f / (1.f + expf(-o));
  float c = sf * c0 + si * tanhf(gg);
  float h = so * tanhf(c);
  out[O_H + b * NH2 + j] = h;
  out[O_C + b * NH2 + j] = c;
  ws[WS_H + b * NH2 + j] = h;
}

// Per-batch h·wa[256:512] plus scalar coverage terms. grid(128) block(32)
__global__ void k_sh(const float* __restrict__ w_align,
                     const float* __restrict__ b_align,
                     const float* __restrict__ w_cov,
                     const float* __restrict__ b_cov,
                     float* ws) {
  int b = blockIdx.x, lane = threadIdx.x;
  float acc = 0.f;
  #pragma unroll
  for (int i = 0; i < 8; ++i) {
    int k = i * 32 + lane;
    acc += ws[WS_H + b * NH2 + k] * w_align[NH2 + k];
  }
  acc = waveReduceSum(acc);
  if (lane == 0) ws[WS_SH + b] = acc;
  if (b == 0) {
    float sc = 0.f, sb = 0.f;
    #pragma unroll
    for (int i = 0; i < 8; ++i) {
      int k = i * 32 + lane;
      float wa = w_align[512 + k];
      sc += w_cov[k] * wa;
      sb += b_cov[k] * wa;
    }
    sc = waveReduceSum(sc);
    sb = waveReduceSum(sb);
    if (lane == 0) { ws[WS_CONST] = sc; ws[WS_CONST + 1] = sb + b_align[0]; }
  }
}

// scores[b,t] = tanh(enc[b,t]·wa[0:256] + s_h[b] + cov[b,t]*s_cov + s_b)
// grid(6400) block(256): one wave per (b,t); 52 MB streaming read.
__global__ void k_scores(const float* __restrict__ enc,
                         const float* __restrict__ cov,
                         const float* __restrict__ w_align,
                         float* ws) {
  int lane = threadIdx.x & 31;
  int gw = blockIdx.x * 8 + (threadIdx.x >> 5);
  int b = gw / NT, t = gw % NT;
  const float* ep = enc + ((size_t)b * NT + t) * NH2 + lane * 8;
  const float* wp = w_align + lane * 8;
  float acc = 0.f;
  #pragma unroll
  for (int i = 0; i < 8; ++i) acc += ep[i] * wp[i];
  acc = waveReduceSum(acc);
  if (lane == 0) {
    float s = tanhf(acc + ws[WS_SH + b] + cov[b * NT + t] * ws[WS_CONST] + ws[WS_CONST + 1]);
    ws[WS_SCORES + b * NT + t] = s;
  }
}

// softmax over T, new_cov, cov_loss. grid(128) block(512)
__global__ void k_softmaxT(const float* __restrict__ cov, float* ws, float* out) {
  __shared__ float red[512];
  int b = blockIdx.x, t = threadIdx.x;
  float s = (t < NT) ? ws[WS_SCORES + b * NT + t] : -3.0e38f;
  red[t] = s; __syncthreads();
  for (int st = 256; st > 0; st >>= 1) {
    if (t < st) red[t] = fmaxf(red[t], red[t + st]);
    __syncthreads();
  }
  float mx = red[0]; __syncthreads();
  float e = (t < NT) ? expf(s - mx) : 0.f;
  red[t] = e; __syncthreads();
  for (int st = 256; st > 0; st >>= 1) {
    if (t < st) red[t] += red[t + st];
    __syncthreads();
  }
  float sm = red[0]; __syncthreads();
  float cl = 0.f;
  if (t < NT) {
    float a = e / sm;
    float cv = cov[b * NT + t];
    out[O_ATT + b * NT + t] = a;
    out[O_NCOV + b * NT + t] = cv + a;
    cl = fminf(a, cv);
  }
  red[t] = cl; __syncthreads();
  for (int st = 256; st > 0; st >>= 1) {
    if (t < st) red[t] += red[t + st];
    __syncthreads();
  }
  if (t == 0) atomicAdd(out + O_CL, red[0]);
}

// context[b,e] = sum_t attn[b,t]*enc[b,t,e]. grid(128) block(256)
__global__ void k_context(const float* __restrict__ enc, float* ws, const float* __restrict__ out) {
  __shared__ float at[NT];
  int b = blockIdx.x, tid = threadIdx.x;
  for (int i = tid; i < NT; i += 256) at[i] = out[O_ATT + b * NT + i];
  __syncthreads();
  const float* ep = enc + (size_t)b * NT * NH2 + tid;
  float acc = 0.f;
  for (int t = 0; t < NT; ++t) acc += at[t] * ep[(size_t)t * NH2];
  ws[WS_CTX + b * NH2 + tid] = acc;
}

// pgen[b] = sigmoid([context,h,emb]·w_gen + b_gen). grid(128) block(32)
__global__ void k_pgen(const int* __restrict__ ids,
                       const float* __restrict__ emb_table,
                       const float* __restrict__ w_gen,
                       const float* __restrict__ b_gen,
                       float* ws) {
  int b = blockIdx.x, lane = threadIdx.x;
  float acc = 0.f;
  for (int i = 0; i < 20; ++i) {
    int idx = i * 32 + lane;  // 0..639
    float x;
    if (idx < 256)      x = ws[WS_CTX + b * NH2 + idx];
    else if (idx < 512) x = ws[WS_H + b * NH2 + (idx - 256)];
    else                x = emb_table[(size_t)ids[b] * NE + (idx - 512)];
    acc += x * w_gen[idx];
  }
  acc = waveReduceSum(acc);
  if (lane == 0) ws[WS_PGEN + b] = 1.f / (1.f + expf(-(acc + b_gen[0])));
}

// fc = tanh([h|context] @ w_fc1.T + b_fc1), stored bf16. grid(16) block(256)
__global__ void k_fc1(const float* __restrict__ w_fc1,
                      const float* __restrict__ b_fc1,
                      float* ws) {
  __shared__ float xs[8][NH4];
  const int bb = blockIdx.x * 8, tid = threadIdx.x;
  for (int i = tid; i < 8 * NH4; i += 256) {
    int lb = i >> 9, e = i & 511;
    xs[lb][e] = (e < 256) ? ws[WS_H + (bb + lb) * NH2 + e]
                          : ws[WS_CTX + (bb + lb) * NH2 + (e - 256)];
  }
  __syncthreads();
  __bf16* fcb = (__bf16*)(ws + WS_FCB);
  for (int q = 0; q < 2; ++q) {
    int j = tid + q * 256;
    float acc[8];
    float bias = b_fc1[j];
    #pragma unroll
    for (int lb = 0; lb < 8; ++lb) acc[lb] = bias;
    const float* wr = w_fc1 + (size_t)j * NH4;
    for (int k = 0; k < NH4; ++k) {
      float w = wr[k];
      #pragma unroll
      for (int lb = 0; lb < 8; ++lb) acc[lb] += w * xs[lb][k];
    }
    #pragma unroll
    for (int lb = 0; lb < 8; ++lb)
      fcb[(size_t)(bb + lb) * NH4 + j] = (__bf16)tanhf(acc[lb]);
  }
}

// --------- dominant kernel: logits = fc(bf16) @ w_fc2.T + b_fc2 ------------
// grid(3125) block(256): block owns a 16-column stripe of V (16 rows of
// w_fc2); each of 8 waves owns a 16-row block of M=128, so the B fragments
// are identical across waves. The stripe (16x512 fp32, 32 KB) is DMA'd into
// LDS once per block by the Tensor Data Mover (TDM) with row-skew padding
// (2 dwords per 256 => pitch 516 dwords => rows land 4 banks apart), then
// converted to bf16 in registers and fed to v_wmma_f32_16x16x32_bf16 with
// two independent accumulator chains.
#define BPITCH 516   // dwords per staged w_fc2 row in LDS

__device__ __forceinline__ void load_frags(const __bf16* __restrict__ arow,
                                           const float* __restrict__ brow,
                                           int k0, int half, v16bf& a, v16bf& b) {
  #pragma unroll
  for (int e = 0; e < 8; ++e) {
    a[e]     = arow[k0 + half * 8 + e];
    a[e + 8] = arow[k0 + 16 + half * 8 + e];
  }
  const int c0 = k0 + half * 8;
  const int c1 = k0 + 16 + half * 8;
  const float* p0 = brow + c0 + 2 * (c0 >> 8);   // TDM pad: +2 dwords per 256
  const float* p1 = brow + c1 + 2 * (c1 >> 8);
  #pragma unroll
  for (int e = 0; e < 8; ++e) { b[e] = (__bf16)p0[e]; b[e + 8] = (__bf16)p1[e]; }
}

__global__ void k_vocab(const float* __restrict__ w_fc2,
                        const float* __restrict__ b_fc2,
                        const float* __restrict__ ws_ro,
                        float* __restrict__ out) {
  __shared__ float bsh[16 * BPITCH];             // 33,024 B staged stripe
  const int lane = threadIdx.x & 31;
  const int wv   = threadIdx.x >> 5;
  const int m0   = wv * 16;
  const int v0   = blockIdx.x * 16;
  const int half = lane >> 4;
  const int l15  = lane & 15;

  // --- TDM: DMA w_fc2[v0 : v0+16, 0:512] (fp32) into LDS, wave 0 only ---
  if (threadIdx.x < 32) {
    unsigned long long ga = (unsigned long long)(uintptr_t)(w_fc2 + (size_t)v0 * NH4);
    unsigned ldsb = (unsigned)(uintptr_t)(void*)&bsh[0];
    v4u g0;
    g0[0] = 1u;                                  // count=1 (valid), no gather
    g0[1] = ldsb;                                // lds_addr (bytes)
    g0[2] = (unsigned)ga;                        // global_addr[31:0]
    g0[3] = (unsigned)((ga >> 32) & 0x01FFFFFFu) | (2u << 30);  // addr[56:32] | type=2
    v8i g1;
    g1[0] = (int)((2u << 16) |                   // data_size = 4 bytes
                  (1u << 20) |                   // pad_enable
                  (7u << 22) |                   // pad_interval: every 256 dwords
                  (1u << 25));                   // pad_amount: 2 dwords
    g1[1] = (int)(512u << 16);                   // tensor_dim0 = 512 (bits 79:48)
    g1[2] = (int)(50000u << 16);                 // tensor_dim1 low16 (bits 111:80)
    g1[3] = (int)((50000u >> 16) | (512u << 16));// tensor_dim1 hi | tile_dim0 = 512
    g1[4] = 16;                                  // tile_dim1 = 16, tile_dim2 = 0
    g1[5] = 512;                                 // tensor_dim0_stride = 512
    g1[6] = 0;
    g1[7] = 0;
    v4i z4 = {0, 0, 0, 0};
#if __clang_major__ >= 23
    v8i z8 = {0, 0, 0, 0, 0, 0, 0, 0};
    __builtin_amdgcn_tensor_load_to_lds(g0, g1, z4, z4, z8, 0);
#else
    __builtin_amdgcn_tensor_load_to_lds(g0, g1, z4, z4, 0);
#endif
    __builtin_amdgcn_s_wait_tensorcnt(0);
  }
  __syncthreads();

  const __bf16* fcb  = (const __bf16*)(ws_ro + WS_FCB);
  const __bf16* arow = fcb + (size_t)(m0 + l15) * NH4;   // this wave's A row
  const float*  brow = &bsh[(size_t)l15 * BPITCH];       // staged B row (LDS)

  v8f acc0 = {0.f, 0.f, 0.f, 0.f, 0.f, 0.f, 0.f, 0.f};
  v8f acc1 = {0.f, 0.f, 0.f, 0.f, 0.f, 0.f, 0.f, 0.f};
  for (int kt = 0; kt < 16; kt += 2) {
    __builtin_prefetch((const void*)(arow + (kt + 2) * 32), 0, 0);
    v16bf a0, b0, a1, b1;
    load_frags(arow, brow, kt * 32, half, a0, b0);
    load_frags(arow, brow, (kt + 1) * 32, half, a1, b1);
    acc0 = __builtin_amdgcn_wmma_f32_16x16x32_bf16(false, a0, false, b0,
                                                   (short)0, acc0, false, false);
    acc1 = __builtin_amdgcn_wmma_f32_16x16x32_bf16(false, a1, false, b1,
                                                   (short)0, acc1, false, false);
  }
  const float bias = b_fc2[v0 + l15];
  #pragma unroll
  for (int r = 0; r < 8; ++r) {
    int row = m0 + half * 8 + r;                 // C/D layout: VGPR r = row r / r+8
    out[O_OUT + (size_t)row * NV + v0 + l15] = acc0[r] + acc1[r] + bias;
  }
}

// row max over V. grid(196,128) block(256)
__global__ void k_rowmax(float* ws, const float* __restrict__ out) {
  __shared__ float red[256];
  int b = blockIdx.y, v = blockIdx.x * 256 + threadIdx.x, t = threadIdx.x;
  float x = (v < NV) ? out[O_OUT + (size_t)b * NV + v] : -3.0e38f;
  red[t] = x; __syncthreads();
  for (int st = 128; st > 0; st >>= 1) {
    if (t < st) red[t] = fmaxf(red[t], red[t + st]);
    __syncthreads();
  }
  if (t == 0) atomicMaxFloat(&ws[WS_RMAX + b], red[0]);
}

// exp in place + row sums. grid(196,128) block(256)
__global__ void k_exp(float* ws, float* out) {
  __shared__ float red[256];
  int b = blockIdx.y, v = blockIdx.x * 256 + threadIdx.x, t = threadIdx.x;
  float e = 0.f;
  if (v < NV) {
    size_t idx = O_OUT + (size_t)b * NV + v;
    e = expf(out[idx] - ws[WS_RMAX + b]);
    out[idx] = e;
  }
  red[t] = e; __syncthreads();
  for (int st = 128; st > 0; st >>= 1) {
    if (t < st) red[t] += red[t + st];
    __syncthreads();
  }
  if (t == 0) atomicAdd(&ws[WS_RSUM + b], red[0]);
}

// scatter attention mass into copy distribution. grid(128) block(512)
__global__ void k_copy(const int* __restrict__ src, float* out) {
  int b = blockIdx.x, t = threadIdx.x;
  if (t < NT) {
    float a = out[O_ATT + b * NT + t];
    int v = src[b * NT + t];
    atomicAdd(out + O_COPY + (size_t)b * NV + v, a);
  }
}

// out = pvocab*pgen + copy*(1-pgen), in place. grid(196,128) block(256)
__global__ void k_final(const float* __restrict__ ws, float* out) {
  int b = blockIdx.y, v = blockIdx.x * 256 + threadIdx.x;
  if (v >= NV) return;
  size_t idx = (size_t)b * NV + v;
  float pg = ws[WS_PGEN + b];
  float pv = out[O_OUT + idx] / ws[WS_RSUM + b];
  out[O_OUT + idx] = pv * pg + out[O_COPY + idx] * (1.f - pg);
}

// ---------------------------------------------------------------------------
extern "C" void kernel_launch(void* const* d_in, const int* in_sizes, int n_in,
                              void* d_out, int out_size, void* d_ws, size_t ws_size,
                              hipStream_t stream) {
  const int*   ids       = (const int*)  d_in[0];
  const float* pre_h     = (const float*)d_in[1];
  const float* pre_c     = (const float*)d_in[2];
  const float* enc       = (const float*)d_in[3];
  const int*   src       = (const int*)  d_in[4];
  const float* cov       = (const float*)d_in[5];
  const float* emb_table = (const float*)d_in[6];
  const float* w_ih      = (const float*)d_in[7];
  const float* w_hh      = (const float*)d_in[8];
  const float* b_ih      = (const float*)d_in[9];
  const float* b_hh      = (const float*)d_in[10];
  const float* w_align   = (const float*)d_in[11];
  const float* b_align   = (const float*)d_in[12];
  const float* w_cov     = (const float*)d_in[13];
  const float* b_cov     = (const float*)d_in[14];
  const float* w_fc1     = (const float*)d_in[15];
  const float* b_fc1     = (const float*)d_in[16];
  const float* w_fc2     = (const float*)d_in[17];
  const float* b_fc2     = (const float*)d_in[18];
  const float* w_gen     = (const float*)d_in[19];
  const float* b_gen     = (const float*)d_in[20];

  float* out = (float*)d_out;
  float* ws  = (float*)d_ws;

  hipMemsetAsync(out + O_COPY, 0, (size_t)NB * NV * sizeof(float), stream);
  k_init<<<1, 256, 0, stream>>>(ws, out);
  k_gates<<<16, 256, 0, stream>>>(ids, emb_table, pre_h, w_ih, w_hh, b_ih, b_hh, ws);
  k_lstm<<<128, 256, 0, stream>>>(pre_c, ws, out);
  k_sh<<<128, 32, 0, stream>>>(w_align, b_align, w_cov, b_cov, ws);
  k_scores<<<6400, 256, 0, stream>>>(enc, cov, w_align, ws);
  k_softmaxT<<<128, 512, 0, stream>>>(cov, ws, out);
  k_context<<<128, 256, 0, stream>>>(enc, ws, out);
  k_pgen<<<128, 32, 0, stream>>>(ids, emb_table, w_gen, b_gen, ws);
  k_fc1<<<16, 256, 0, stream>>>(w_fc1, b_fc1, ws);
  k_vocab<<<3125, 256, 0, stream>>>(w_fc2, b_fc2, ws, out);
  k_rowmax<<<dim3(196, 128), 256, 0, stream>>>(ws, out);
  k_exp<<<dim3(196, 128), 256, 0, stream>>>(ws, out);
  k_copy<<<128, 512, 0, stream>>>(src, out);
  k_final<<<dim3(196, 128), 256, 0, stream>>>(ws, out);
}
